// SingleLayerRGN_38319698215247
// MI455X (gfx1250) — matrix operand order repaired and checked by reading, not compiled, not run on hardware
//
#include <hip/hip_runtime.h>
#include <hip/hip_bf16.h>

// ---------------- problem constants (match reference) ----------------
#define NF_   64
#define EF_   16
#define H_    128
#define OUT_  64
#define T_    4
#define NT_   2
#define KIN_  144          // NF+EF+NF
#define KPAD_ 160          // padded to multiple of 32 for 16x16x32 WMMA

typedef _Float16 half8   __attribute__((ext_vector_type(8)));
typedef _Float16 half16  __attribute__((ext_vector_type(16)));
typedef float    float8  __attribute__((ext_vector_type(8)));
typedef float    float4v __attribute__((ext_vector_type(4)));

// ---- A fragment (16x32 f16) from an f16 row in LDS.
// lane L: M=L&15, kh=L>>4; rowptr already includes +kh*8; kbase = kk*32.
__device__ inline half16 frag_a(const _Float16* rowptr, int kbase) {
    half8 lo = *(const half8*)(rowptr + kbase);
    half8 hi = *(const half8*)(rowptr + kbase + 16);
    half16 a;
#pragma unroll
    for (int i = 0; i < 8; ++i) { a[i] = lo[i]; a[i + 8] = hi[i]; }
    return a;
}

// ---- A fragment from an f32 row in LDS (convert at use; amortized over 8 WMMAs)
__device__ inline half16 frag_a_f32(const float* rowptr, int kbase) {
    const float4v* p = (const float4v*)(rowptr + kbase);
    float4v x0 = p[0], x1 = p[1];
    const float4v* q = (const float4v*)(rowptr + kbase + 16);
    float4v y0 = q[0], y1 = q[1];
    half16 a;
#pragma unroll
    for (int i = 0; i < 4; ++i) {
        a[i]      = (_Float16)x0[i];
        a[4 + i]  = (_Float16)x1[i];
        a[8 + i]  = (_Float16)y0[i];
        a[12 + i] = (_Float16)y1[i];
    }
    return a;
}

// ---- B fragment (32x16 f16): lane L: n=L&15, ksel=L>>4.
// p points at W_t[n][kk*32 + ksel*8] (weights stored [N][Kpad] f16): halfs p[0..7] and p[16..23].
__device__ inline half16 frag_b(const _Float16* p) {
    half8 lo = *(const half8*)p;
    half8 hi = *(const half8*)(p + 16);
    half16 b;
#pragma unroll
    for (int i = 0; i < 8; ++i) { b[i] = lo[i]; b[i + 8] = hi[i]; }
    return b;
}

// ---------------- init: zero agg, cnt, hist+cursor ----------------
__global__ void k_init(float* agg, float* cnt, int* hist8, int N) {
    int i = blockIdx.x * 256 + threadIdx.x;
    if (i < N * OUT_) agg[i] = 0.0f;
    if (i < N)        cnt[i] = 0.0f;
    if (i < 8)        hist8[i] = 0;     // hist[0..3], cursor[4..7]
}

// ---------------- weight convert + transpose to [N][K] f16 ----------------
__global__ void k_prep_weights(const float* mW1, const float* mW2,
                               const float* uW1, const float* uW2,
                               _Float16* Wt1, _Float16* Wt2,
                               _Float16* Ut1, _Float16* Ut2) {
    const int n1 = T_ * H_ * KPAD_;      // 4*128*160
    const int n2 = T_ * OUT_ * H_;       // 4*64*128
    const int n3 = NT_ * H_ * NF_;       // 2*128*64
    const int n4 = NT_ * OUT_ * H_;      // 2*64*128
    int i = blockIdx.x * 256 + threadIdx.x;
    if (i < n1) {
        int k = i % KPAD_; int rem = i / KPAD_; int n = rem % H_; int t = rem / H_;
        Wt1[i] = (k < KIN_) ? (_Float16)mW1[(t * KIN_ + k) * H_ + n] : (_Float16)0.0f;
    } else if (i < n1 + n2) {
        int j = i - n1;
        int k = j % H_; int rem = j / H_; int n = rem % OUT_; int t = rem / OUT_;
        Wt2[j] = (_Float16)mW2[(t * H_ + k) * OUT_ + n];
    } else if (i < n1 + n2 + n3) {
        int j = i - n1 - n2;
        int k = j % NF_; int rem = j / NF_; int n = rem % H_; int t = rem / H_;
        Ut1[j] = (_Float16)uW1[(t * NF_ + k) * H_ + n];
    } else if (i < n1 + n2 + n3 + n4) {
        int j = i - n1 - n2 - n3;
        int k = j % H_; int rem = j / H_; int n = rem % OUT_; int t = rem / OUT_;
        Ut2[j] = (_Float16)uW2[(t * H_ + k) * OUT_ + n];
    }
}

// ---------------- edge-type binning ----------------
__global__ void k_hist(const int* etype, int* hist8, int E) {
    int e = blockIdx.x * 256 + threadIdx.x;
    if (e < E) atomicAdd(&hist8[etype[e]], 1);
}
__global__ void k_scan(const int* hist8, int* offs) {
    if (threadIdx.x == 0) {
        int a = 0;
#pragma unroll
        for (int t = 0; t < T_; ++t) { offs[t] = a; a += hist8[t]; }
    }
}
__global__ void k_scatter(const int* etype, const int* offs, int* cursor,
                          int* perm, int E) {
    int e = blockIdx.x * 256 + threadIdx.x;
    if (e < E) {
        int t = etype[e];
        int pos = offs[t] + atomicAdd(&cursor[t], 1);
        perm[pos] = e;
    }
}

// LDS layout (single arena so W1 sits at LDS offset 0 -> 16-bit ds immediates)
#define SM_W1    0                         // 128*160 f16 = 40960 B
#define SM_XS    (H_ * KPAD_ * 2)          // 40960: 4 x (16*160 f32) = 40960 B
#define SM_HS    (SM_XS + 4 * 16 * KPAD_ * 4)  // 81920: 4 x (16*128 f16) = 16384 B
#define SM_EID   (SM_HS + 4 * 16 * H_ * 2)     // 98304
#define SM_SRC   (SM_EID + 64 * 4)
#define SM_DST   (SM_SRC + 64 * 4)
#define SM_TOTAL (SM_DST + 64 * 4)             // 99072 B

// ---------------- edge message MLP: 64 edges / block, 4 waves ----------------
// Wave w owns edges [tile*64 + w*16, +16): its own M-tile across all N-tiles.
__global__ __launch_bounds__(128, 4) void k_edge_mlp(
    const float* __restrict__ nf, const float* __restrict__ ef,
    const _Float16* __restrict__ Wt1, const _Float16* __restrict__ Wt2,
    const float* __restrict__ b1, const float* __restrict__ b2,
    const int* __restrict__ perm, const int* __restrict__ hist8,
    const int* __restrict__ offs,
    const int* __restrict__ esrc, const int* __restrict__ edst,
    float* __restrict__ agg, float* __restrict__ cnt) {
    __shared__ __align__(16) char smem[SM_TOTAL];
    _Float16* w1s  = (_Float16*)(smem + SM_W1);
    int* s_eid = (int*)(smem + SM_EID);
    int* s_src = (int*)(smem + SM_SRC);
    int* s_dst = (int*)(smem + SM_DST);

    // locate (type, tile) — uniform across the workgroup
    int g = blockIdx.x;
    int ty = -1, tile = 0, binbase = 0, bincnt = 0, acc = 0;
#pragma unroll
    for (int t = 0; t < T_; ++t) {
        int c = hist8[t];
        int nt = (c + 63) >> 6;
        if (ty < 0 && g < acc + nt) { ty = t; tile = g - acc; binbase = offs[t]; bincnt = c; }
        acc += nt;
    }
    if (ty < 0) return;                    // uniform exit

    int tid = threadIdx.x, lane = tid & 31, wv = tid >> 5;
    int rowbase = wv * 16;
    float* xw = (float*)(smem + SM_XS) + wv * 16 * KPAD_;
    _Float16* hw = (_Float16*)(smem + SM_HS) + wv * 16 * H_;

    // per-edge staging: lanes 0..15 of each wave handle its own 16 edges
    if (lane < 16) {
        int p = tile * 64 + rowbase + lane;
        int e = (p < bincnt) ? perm[binbase + p] : -1;
        s_eid[rowbase + lane] = e;
        int s = 0, d = 0;
        if (e >= 0) { s = esrc[e]; d = edst[e]; atomicAdd(&cnt[d], 1.0f); }
        s_src[rowbase + lane] = s;
        s_dst[rowbase + lane] = d;
    }

    // stage W1[type] into LDS (whole block cooperates)
    {
        const half8* src = (const half8*)(Wt1 + (size_t)ty * H_ * KPAD_);
        half8* dst = (half8*)w1s;
        for (int i = tid; i < H_ * KPAD_ / 8; i += 128) dst[i] = src[i];
    }

    // async gather: 16 rows x 36 b128 chunks (f32) -> xw
    {
#pragma unroll
        for (int it = 0; it < 18; ++it) {
            int idx = it * 32 + lane;                 // 0..575
            int row = idx / 36, c4 = idx - row * 36;
            int col = c4 * 4;
            int e = s_eid[rowbase + row];
            if (e >= 0) {
                const float* gp;
                if (col < NF_)            gp = nf + (size_t)s_src[rowbase + row] * NF_ + col;
                else if (col < NF_ + EF_) gp = ef + (size_t)e * EF_ + (col - NF_);
                else                      gp = nf + (size_t)s_dst[rowbase + row] * NF_ + (col - NF_ - EF_);
                unsigned ldsa = (unsigned)(uintptr_t)(&xw[row * KPAD_ + col]);
                unsigned long long ga = (unsigned long long)(uintptr_t)gp;
                asm volatile("global_load_async_to_lds_b128 %0, %1, off"
                             :: "v"(ldsa), "v"(ga) : "memory");
            }
        }
        // zero K-pad cols 144..159 (4 float4 chunks per row)
        for (int i = lane; i < 64; i += 32) {
            int row = i >> 2, c4 = i & 3;
            *(float4v*)&xw[row * KPAD_ + KIN_ + c4 * 4] = (float4v){0.f, 0.f, 0.f, 0.f};
        }
        // tail waves only: zero whole invalid rows (wave-uniform branch)
        if (tile * 64 + rowbase + 16 > bincnt) {
            for (int i = lane; i < 16 * KPAD_; i += 32) {
                int row = i / KPAD_;
                if (s_eid[rowbase + row] < 0) xw[i] = 0.0f;
            }
        }
        asm volatile("s_wait_asynccnt 0" ::: "memory");
    }
    __syncthreads();   // W1 in LDS visible to all waves

    int m = lane & 15, kh = lane >> 4;

    // ---- GEMM1: [16x160](f32->f16) x [160x128], all 8 N-tiles per wave ----
    // Real loop over kk (bounded register pressure); inside one iteration:
    // A fragment for kk+1 prefetched, B fragments ping-pong double-buffered.
    const float* ap = &xw[m * KPAD_ + kh * 8];
    const _Float16* bp = w1s + m * KPAD_ + kh * 8;      // + nt*16*KPAD_ (immediates)
    float8 c1v[8];
#pragma unroll
    for (int nt = 0; nt < 8; ++nt) c1v[nt] = (float8){};

    half16 a_cur = frag_a_f32(ap, 0);
#pragma unroll 1
    for (int kk = 0; kk < KPAD_ / 32; ++kk) {
        half16 a_nxt;
        if (kk + 1 < KPAD_ / 32) a_nxt = frag_a_f32(ap, 32);
        half16 bc = frag_b(bp);                          // nt = 0
#pragma unroll
        for (int nt = 0; nt < 8; ++nt) {
            half16 bn;
            if (nt < 7) bn = frag_b(bp + (nt + 1) * 16 * KPAD_);
            c1v[nt] = __builtin_amdgcn_wmma_f32_16x16x32_f16(false, a_cur, false, bc,
                                                             (short)0, c1v[nt], false, false);
            if (nt < 7) bc = bn;
        }
        if (kk + 1 < KPAD_ / 32) a_cur = a_nxt;
        ap += 32;
        bp += 32;
    }
    // bias + relu -> hw (wave-private; no block barrier needed)
#pragma unroll
    for (int nt = 0; nt < 8; ++nt) {
        int ncol = nt * 16 + m;
        float bias = b1[ty * H_ + ncol];
#pragma unroll
        for (int r = 0; r < 8; ++r) {
            float v = c1v[nt][r] + bias;
            hw[(r + kh * 8) * H_ + ncol] = (_Float16)(v > 0.0f ? v : 0.0f);
        }
    }

    // ---- GEMM2: [16x128] x [128x64], 4 N-tiles per wave; W2 from global (L1-hot) ----
    const _Float16* arow2 = &hw[m * H_ + kh * 8];
    const _Float16* b2base = Wt2 + ((size_t)ty * OUT_ + m) * H_ + kh * 8;
    float8 c2v[4];
#pragma unroll
    for (int nt = 0; nt < 4; ++nt) c2v[nt] = (float8){};

    half16 a2_cur = frag_a(arow2, 0);
#pragma unroll
    for (int kk = 0; kk < H_ / 32; ++kk) {
        half16 a2_nxt;
        if (kk < H_ / 32 - 1) a2_nxt = frag_a(arow2, (kk + 1) * 32);
        half16 bc = frag_b(b2base + kk * 32);            // nt = 0
#pragma unroll
        for (int nt = 0; nt < 4; ++nt) {
            half16 bn;
            if (nt < 3) bn = frag_b(b2base + (size_t)(nt + 1) * 16 * H_ + kk * 32);
            c2v[nt] = __builtin_amdgcn_wmma_f32_16x16x32_f16(false, a2_cur, false, bc,
                                                             (short)0, c2v[nt], false, false);
            if (nt < 3) bc = bn;
        }
        if (kk < H_ / 32 - 1) a2_cur = a2_nxt;
    }
    // scatter-add messages into agg (value-masked, never thread-masked at WMMA)
#pragma unroll
    for (int nt = 0; nt < 4; ++nt) {
        int ncol = nt * 16 + m;
        float bias2 = b2[ty * OUT_ + ncol];
#pragma unroll
        for (int r = 0; r < 8; ++r) {
            int row = rowbase + r + kh * 8;
            if (s_eid[row] >= 0)
                atomicAdd(&agg[(size_t)s_dst[row] * OUT_ + ncol], c2v[nt][r] + bias2);
        }
    }
}

// ---------------- node update MLP: 16 nodes / block, 4 waves ----------------
__global__ __launch_bounds__(128) void k_node_mlp(
    const float* __restrict__ agg, const float* __restrict__ cnt,
    const _Float16* __restrict__ Ut1, const _Float16* __restrict__ Ut2,
    const float* __restrict__ ub1, const float* __restrict__ ub2,
    const int* __restrict__ ntype, float* __restrict__ out, int N) {
    __shared__ __align__(16) _Float16 as[16 * NF_];
    __shared__ __align__(16) _Float16 hs[16 * H_];
    __shared__ int s_nt[16];

    int tid = threadIdx.x;
    int node0 = blockIdx.x * 16;
    if (tid < 16) {
        int nd = node0 + tid;
        s_nt[tid] = (nd < N) ? ntype[nd] : -1;
    }
    for (int i = tid; i < 16 * NF_; i += 128) {
        int row = i >> 6, col = i & 63;
        int nd = node0 + row;
        float v = 0.0f;
        if (nd < N) {
            float c = cnt[nd];
            if (c < 1.0f) c = 1.0f;
            v = agg[(size_t)nd * OUT_ + col] / c;
        }
        as[i] = (_Float16)v;
    }
    __syncthreads();

    int lane = tid & 31, wv = tid >> 5;
    int m = lane & 15, kh = lane >> 4;
    int ncol2 = wv * 16 + m;
    float outacc[8];
#pragma unroll
    for (int r = 0; r < 8; ++r) outacc[r] = 0.0f;

    for (int nt = 0; nt < NT_; ++nt) {
        const _Float16* arow = &as[m * NF_ + kh * 8];
        const _Float16* br0  = &Ut1[((size_t)nt * H_ + (wv * 2 + 0) * 16 + m) * NF_ + kh * 8];
        const _Float16* br1  = &Ut1[((size_t)nt * H_ + (wv * 2 + 1) * 16 + m) * NF_ + kh * 8];
        float8 c0 = {}, c1 = {};
#pragma unroll
        for (int kk = 0; kk < NF_ / 32; ++kk) {
            half16 a  = frag_a(arow, kk * 32);
            half16 bA = frag_b(br0 + kk * 32);
            c0 = __builtin_amdgcn_wmma_f32_16x16x32_f16(false, a, false, bA, (short)0, c0, false, false);
            half16 bB = frag_b(br1 + kk * 32);
            c1 = __builtin_amdgcn_wmma_f32_16x16x32_f16(false, a, false, bB, (short)0, c1, false, false);
        }
#pragma unroll
        for (int p = 0; p < 2; ++p) {
            int ncol = (wv * 2 + p) * 16 + m;
            float bias = ub1[nt * H_ + ncol];
            float8 cc = p ? c1 : c0;
#pragma unroll
            for (int r = 0; r < 8; ++r) {
                float v = cc[r] + bias;
                hs[(r + kh * 8) * H_ + ncol] = (_Float16)(v > 0.0f ? v : 0.0f);
            }
        }
        __syncthreads();

        const _Float16* arow2 = &hs[m * H_ + kh * 8];
        const _Float16* br2   = &Ut2[((size_t)nt * OUT_ + ncol2) * H_ + kh * 8];
        float8 c2 = {};
#pragma unroll
        for (int kk = 0; kk < H_ / 32; ++kk) {
            half16 a = frag_a(arow2, kk * 32);
            half16 b = frag_b(br2 + kk * 32);
            c2 = __builtin_amdgcn_wmma_f32_16x16x32_f16(false, a, false, b, (short)0, c2, false, false);
        }
        float bias2 = ub2[nt * OUT_ + ncol2];
#pragma unroll
        for (int r = 0; r < 8; ++r) {
            int row = r + kh * 8;
            if (s_nt[row] == nt) outacc[r] = c2[r] + bias2;
        }
        __syncthreads();   // before next type overwrites hs
    }

#pragma unroll
    for (int r = 0; r < 8; ++r) {
        int nd = node0 + r + kh * 8;
        if (nd < N) out[(size_t)nd * OUT_ + ncol2] = outacc[r];
    }
}

// ---------------- host launch ----------------
extern "C" void kernel_launch(void* const* d_in, const int* in_sizes, int n_in,
                              void* d_out, int out_size, void* d_ws, size_t ws_size,
                              hipStream_t stream) {
    const float* nf   = (const float*)d_in[0];
    const float* ef   = (const float*)d_in[1];
    const float* mW1  = (const float*)d_in[2];
    const float* mb1  = (const float*)d_in[3];
    const float* mW2  = (const float*)d_in[4];
    const float* mb2  = (const float*)d_in[5];
    const float* uW1  = (const float*)d_in[6];
    const float* ub1  = (const float*)d_in[7];
    const float* uW2  = (const float*)d_in[8];
    const float* ub2  = (const float*)d_in[9];
    const int* esrc   = (const int*)d_in[10];
    const int* edst   = (const int*)d_in[11];
    const int* etype  = (const int*)d_in[12];
    const int* ntype  = (const int*)d_in[13];

    const int N = in_sizes[0] / NF_;
    const int E = in_sizes[10];

    // workspace carve-up (256B aligned slices)
    char* ws = (char*)d_ws;
    size_t off = 0;
    auto take = [&](size_t bytes) {
        char* p = ws + off;
        off = (off + bytes + 255) & ~(size_t)255;
        return p;
    };
    _Float16* Wt1 = (_Float16*)take((size_t)T_ * H_ * KPAD_ * 2);
    _Float16* Wt2 = (_Float16*)take((size_t)T_ * OUT_ * H_ * 2);
    _Float16* Ut1 = (_Float16*)take((size_t)NT_ * H_ * NF_ * 2);
    _Float16* Ut2 = (_Float16*)take((size_t)NT_ * OUT_ * H_ * 2);
    float* agg    = (float*)take((size_t)N * OUT_ * 4);
    float* cnt    = (float*)take((size_t)N * 4);
    int* perm     = (int*)take((size_t)E * 4);
    int* hist8    = (int*)take(8 * 4);     // hist[4] + cursor[4]
    int* offs     = (int*)take(4 * 4);
    int* cursor   = hist8 + 4;

    // 1) zero agg/cnt/hist/cursor
    {
        int total = N * OUT_;
        k_init<<<(total + 255) / 256, 256, 0, stream>>>(agg, cnt, hist8, N);
    }
    // 2) convert + transpose weights to f16 [N][K]
    {
        int total = T_ * H_ * KPAD_ + T_ * OUT_ * H_ + NT_ * H_ * NF_ + NT_ * OUT_ * H_;
        k_prep_weights<<<(total + 255) / 256, 256, 0, stream>>>(
            mW1, mW2, uW1, uW2, Wt1, Wt2, Ut1, Ut2);
    }
    // 3) bin edges by type
    k_hist<<<(E + 255) / 256, 256, 0, stream>>>(etype, hist8, E);
    k_scan<<<1, 32, 0, stream>>>(hist8, offs);
    k_scatter<<<(E + 255) / 256, 256, 0, stream>>>(etype, offs, cursor, perm, E);
    // 4) edge MLP + scatter-add (64 edges / block)
    {
        int maxBlocks = (E + 63) / 64 + T_;
        k_edge_mlp<<<maxBlocks, 128, 0, stream>>>(
            nf, ef, Wt1, Wt2, mb1, mb2, perm, hist8, offs, esrc, edst, agg, cnt);
    }
    // 5) node MLP
    k_node_mlp<<<(N + 15) / 16, 128, 0, stream>>>(
        agg, cnt, Ut1, Ut2, ub1, ub2, ntype, (float*)d_out, N);
}
